// HybridLOD2Model_22213570855532
// MI455X (gfx1250) — matrix operand order repaired and checked by reading, not compiled, not run on hardware
//
#include <hip/hip_runtime.h>
#include <math.h>
#include <stdint.h>

// ---------------------------------------------------------------------------
// HybridLOD2 point-cloud model forward on gfx1250 (wave32, WMMA).
// All GEMM-shaped math -> v_wmma_f32_16x16x32_f16 (f16 in, f32 acc).
// GEMM has a guard-free fast path for tile-aligned shapes (all transformer
// GEMMs) and a clamped-select guarded path for ragged SA shapes.
// Attention: flash-style streaming softmax; K/V tiles staged in LDS once per
// block via gfx1250 async global->LDS (builtin present on this toolchain).
// Input flattening assumed depth-first in dict-insertion order:
//   0: xyz
//   1..48: sa[stage][layer]{w,b,g,be}  stages {1,13,25,37}+4*layer
//   49,50: pos_lin{w,b}   51,52: input_proj{w,b}
//   53+12*l .. : tf layer l {wqkv,bqkv,wo,bo,lin1.w,lin1.b,lin2.w,lin2.b,n1g,n1b,n2g,n2b}
//   101..104: pool1{w,b} pool2{w,b}
//   105..110: f_wqkv,f_bqkv,f_wo,f_bo,fng,fnb
//   111..114: fc1{w,b} fc2{w,b}
//   115..118: cl1{w,b} bn1g,bn1b   119..122: cl2{w,b} bn2g,bn2b   123,124: cl3{w,b}
// ---------------------------------------------------------------------------

#define BATCH  4
#define NPTS   16384
#define TOK    2048
#define DMODEL 512
#define NHEADS 8
#define HDIM   64
#define FFNDIM 2048
#define EPSV   1e-5f

typedef _Float16 v16h __attribute__((ext_vector_type(16)));
typedef _Float16 v8h  __attribute__((ext_vector_type(8)));
typedef float    v8f  __attribute__((ext_vector_type(8)));
typedef int      aint4 __attribute__((vector_size(16)));   // matches builtin V4i

struct FragH { union { v16h v; v8h q[2]; _Float16 h[16]; }; };
struct FragF { union { v8f  v; float    f[8];  }; };
struct Pack8 { union { v8h  v; _Float16 h[8];  }; };

// ---- gfx1250 async global->LDS copy (builtin verified present) -------------
#if defined(__has_builtin)
#if __has_builtin(__builtin_amdgcn_global_load_async_to_lds_b128)
#define HAVE_ASYNC_LDS 1
#endif
#endif

#define AS1 __attribute__((address_space(1)))
#define AS3 __attribute__((address_space(3)))

__device__ __forceinline__ void async_cp16(const float* g, float* l) {
#ifdef HAVE_ASYNC_LDS
  __builtin_amdgcn_global_load_async_to_lds_b128(
      (AS1 aint4*)(uintptr_t)g, (AS3 aint4*)(unsigned)(uintptr_t)l, 0, 0);
#else
  *(float4*)l = *(const float4*)g;
#endif
}
__device__ __forceinline__ void async_fence_block() {
#ifdef HAVE_ASYNC_LDS
#if __has_builtin(__builtin_amdgcn_s_wait_asynccnt)
  __builtin_amdgcn_s_wait_asynccnt(0);
#else
  asm volatile("s_wait_asynccnt 0x0" ::: "memory");
#endif
#endif
  __syncthreads();
}

// ---------------------------------------------------------------------------
// FPS: one workgroup per batch, running min-distance array lives in LDS.
// ---------------------------------------------------------------------------
__global__ void fps_kernel(const float* __restrict__ xyz, int* __restrict__ out,
                           int N, int S) {
  extern __shared__ float dist[];
  __shared__ float rv[256];
  __shared__ int   ri[256];
  __shared__ int   sFar;
  int b = blockIdx.x, tid = threadIdx.x;
  const float* xb = xyz + (long long)b * N * 3;
  for (int j = tid; j < N; j += 256) dist[j] = 1e10f;
  if (tid == 0) sFar = 0;
  __syncthreads();
  for (int i = 0; i < S; i++) {
    int far = sFar;
    if (tid == 0) out[b * S + i] = far;
    float cx = xb[far * 3 + 0], cy = xb[far * 3 + 1], cz = xb[far * 3 + 2];
    float bv = -1.f; int bi = 0;
    for (int j = tid; j < N; j += 256) {
      float dx = xb[j * 3 + 0] - cx, dy = xb[j * 3 + 1] - cy, dz = xb[j * 3 + 2] - cz;
      float d = dx * dx + dy * dy + dz * dz;
      float nd = fminf(dist[j], d);
      dist[j] = nd;
      if (nd > bv) { bv = nd; bi = j; }
    }
    rv[tid] = bv; ri[tid] = bi;
    __syncthreads();
    for (int s2 = 128; s2 > 0; s2 >>= 1) {
      if (tid < s2 && rv[tid + s2] > rv[tid]) { rv[tid] = rv[tid + s2]; ri[tid] = ri[tid + s2]; }
      __syncthreads();
    }
    if (tid == 0) sFar = ri[0];
    __syncthreads();
  }
}

__global__ void gather_kernel(const float* __restrict__ src, const int* __restrict__ idx,
                              float* __restrict__ dst, int N, int S, int C) {
  long long t = (long long)blockIdx.x * 256 + threadIdx.x;
  long long tot = (long long)BATCH * S * C;
  if (t >= tot) return;
  int c = (int)(t % C);
  long long bs = t / C;
  int b = (int)(bs / S), s = (int)(bs % S);
  int j = idx[b * S + s];
  dst[t] = src[((long long)b * N + j) * C + c];
}

// first-K in-radius indices in ascending order; pad with first (matches reference)
__global__ void ball_query_kernel(const float* __restrict__ xyz, const float* __restrict__ nxyz,
                                  int* __restrict__ idx, int N, int S, int K, float r2) {
  int t = blockIdx.x * 256 + threadIdx.x;
  if (t >= BATCH * S) return;
  int b = t / S;
  const float* xb = xyz + (long long)b * N * 3;
  float cx = nxyz[(long long)t * 3 + 0], cy = nxyz[(long long)t * 3 + 1], cz = nxyz[(long long)t * 3 + 2];
  int* o = idx + (long long)t * K;
  int cnt = 0;
  for (int j = 0; j < N && cnt < K; j++) {
    float dx = xb[j * 3] - cx, dy = xb[j * 3 + 1] - cy, dz = xb[j * 3 + 2] - cz;
    if (dx * dx + dy * dy + dz * dz <= r2) o[cnt++] = j;
  }
  int pad = (cnt > 0) ? o[0] : 0;
  for (; cnt < K; cnt++) o[cnt] = pad;
}

__global__ void group_kernel(const float* __restrict__ xyz, const float* __restrict__ nxyz,
                             const float* __restrict__ feats, const int* __restrict__ idx,
                             float* __restrict__ g, int N, int S, int K, int Cf) {
  long long t = (long long)blockIdx.x * 256 + threadIdx.x;
  long long tot = (long long)BATCH * S * K;
  if (t >= tot) return;
  long long bs = t / K;
  int b = (int)(bs / S);
  int j = idx[t];
  int C = 3 + Cf;
  float* o = g + t * C;
  const float* pj = xyz + ((long long)b * N + j) * 3;
  const float* pc = nxyz + bs * 3;
  o[0] = pj[0] - pc[0]; o[1] = pj[1] - pc[1]; o[2] = pj[2] - pc[2];
  if (Cf > 0) {
    const float* fj = feats + ((long long)b * N + j) * Cf;
    for (int c = 0; c < Cf; c++) o[3 + c] = fj[c];
  }
}

__device__ __forceinline__ void cvt4h(Pack8& p, int o, float4 f) {
  p.h[o + 0] = (_Float16)f.x; p.h[o + 1] = (_Float16)f.y;
  p.h[o + 2] = (_Float16)f.z; p.h[o + 3] = (_Float16)f.w;
}

// ---------------------------------------------------------------------------
// WMMA tiled GEMM:  C[M,N] = A[M,K] * W[N,K]^T (+bias)(+relu)
// Block: 256 thr = 8 waves, block tile 64(M) x 128(N), 32x32 per wave.
// GUARD=false: shapes with M%64==0, N%128==0, K%32==0 -> no bounds checks,
// aligned float4 global loads. GUARD=true: branchless clamped staging.
// Every fragment = two contiguous 16B ds_load_b128:
//   sA [row][k]  row stride 80B (16B aligned, conflict-free strided reads)
//   sBt[n][k]    (W consumed untransposed; B frag is k=e+hi*16 at fixed n)
// ---------------------------------------------------------------------------
template <bool GUARD>
__global__ __launch_bounds__(256) void gemm_kernel(const float* __restrict__ A,
                                                   const float* __restrict__ W,
                                                   const float* __restrict__ bias,
                                                   float* __restrict__ C,
                                                   int M, int N, int K, int act) {
  __shared__ _Float16 sA[64][40];
  __shared__ _Float16 sBt[128][40];
  int tid = threadIdx.x, lane = tid & 31, wid = tid >> 5;
  int l15 = lane & 15, hi = (lane >> 4) & 1;
  int Mblk = blockIdx.y * 64, Nblk = blockIdx.x * 128;
  int wm = (wid & 1) * 32, wn = (wid >> 1) * 32;

  // staging coordinates (fixed per thread)
  int ar = tid >> 2, aseg = (tid & 3) * 8;          // A: 64 rows x 4 segs
  int agm = Mblk + ar;
  long long arow = (long long)((!GUARD || agm < M) ? agm : 0) * K;
  int bn0 = tid >> 2, bn1 = 64 + (tid >> 2);        // B: 128 n x 4 segs, 2 tasks
  int bseg = (tid & 3) * 8;
  int gn0 = Nblk + bn0, gn1 = Nblk + bn1;
  long long brow0 = (long long)((!GUARD || gn0 < N) ? gn0 : 0) * K;
  long long brow1 = (long long)((!GUARD || gn1 < N) ? gn1 : 0) * K;

  FragF acc[2][2];
#pragma unroll
  for (int i = 0; i < 2; i++)
#pragma unroll
    for (int j = 0; j < 2; j++)
#pragma unroll
      for (int r = 0; r < 8; r++) acc[i][j].f[r] = 0.f;

  for (int kk = 0; kk < K; kk += 32) {
    Pack8 pa, pb0, pb1;
    if (GUARD) {
#pragma unroll
      for (int j = 0; j < 8; j++) {
        int k = kk + aseg + j;
        bool ok = (agm < M) && (k < K);
        float x = A[ok ? (arow + k) : 0];
        pa.h[j] = ok ? (_Float16)x : (_Float16)0.f;
      }
#pragma unroll
      for (int j = 0; j < 8; j++) {
        int k = kk + bseg + j;
        bool ok = (gn0 < N) && (k < K);
        float x = W[ok ? (brow0 + k) : 0];
        pb0.h[j] = ok ? (_Float16)x : (_Float16)0.f;
      }
#pragma unroll
      for (int j = 0; j < 8; j++) {
        int k = kk + bseg + j;
        bool ok = (gn1 < N) && (k < K);
        float x = W[ok ? (brow1 + k) : 0];
        pb1.h[j] = ok ? (_Float16)x : (_Float16)0.f;
      }
    } else {
      const float* ap  = A + arow  + kk + aseg;   // 32B aligned (K%32==0)
      const float* wp0 = W + brow0 + kk + bseg;
      const float* wp1 = W + brow1 + kk + bseg;
      cvt4h(pa,  0, *(const float4*)(ap));
      cvt4h(pa,  4, *(const float4*)(ap + 4));
      cvt4h(pb0, 0, *(const float4*)(wp0));
      cvt4h(pb0, 4, *(const float4*)(wp0 + 4));
      cvt4h(pb1, 0, *(const float4*)(wp1));
      cvt4h(pb1, 4, *(const float4*)(wp1 + 4));
    }
    *(v8h*)&sA[ar][aseg]   = pa.v;
    *(v8h*)&sBt[bn0][bseg] = pb0.v;
    *(v8h*)&sBt[bn1][bseg] = pb1.v;
    __syncthreads();

    int kbA = hi ? 16 : 0, kbB = hi ? 16 : 0;
    FragH a[2], bf[2];
#pragma unroll
    for (int mi = 0; mi < 2; mi++) {
      int row = wm + mi * 16 + l15;
      a[mi].q[0] = *(const v8h*)&sA[row][kbA];        // e0..7  -> k kbA..kbA+7
      a[mi].q[1] = *(const v8h*)&sA[row][kbA + 8];    // e8..15 -> k kbA+8..+15
    }
#pragma unroll
    for (int ni = 0; ni < 2; ni++) {
      int col = wn + ni * 16 + l15;
      bf[ni].q[0] = *(const v8h*)&sBt[col][kbB];      // e0..7  -> k kbB..kbB+7
      bf[ni].q[1] = *(const v8h*)&sBt[col][kbB + 8];  // e8..15 -> k kbB+8..+15
    }
#pragma unroll
    for (int mi = 0; mi < 2; mi++)
#pragma unroll
      for (int ni = 0; ni < 2; ni++)
        acc[mi][ni].v = __builtin_amdgcn_wmma_f32_16x16x32_f16(
            false, a[mi].v, false, bf[ni].v, (short)0, acc[mi][ni].v, false, false);
    __syncthreads();
  }

#pragma unroll
  for (int mi = 0; mi < 2; mi++)
#pragma unroll
    for (int ni = 0; ni < 2; ni++) {
      int gcol = Nblk + wn + ni * 16 + l15;
      float bval = (bias ? bias[(!GUARD || gcol < N) ? gcol : 0] : 0.f);
#pragma unroll
      for (int r = 0; r < 8; r++) {
        int grow = Mblk + wm + mi * 16 + r + (hi ? 8 : 0);
        if (!GUARD || (grow < M && gcol < N)) {
          float v = acc[mi][ni].f[r] + bval;
          if (act) v = fmaxf(v, 0.f);
          C[(long long)grow * N + gcol] = v;
        }
      }
    }
}

// per-channel mean/var over M rows (BN over all-but-last axes)
__global__ void bn_stats_kernel(const float* __restrict__ x, float* __restrict__ stats,
                                int M, int C) {
  int c = blockIdx.x, tid = threadIdx.x;
  __shared__ float s1[256], s2[256];
  float a = 0.f, q = 0.f;
  for (int m = tid; m < M; m += 256) { float v = x[(long long)m * C + c]; a += v; q += v * v; }
  s1[tid] = a; s2[tid] = q;
  __syncthreads();
  for (int s = 128; s > 0; s >>= 1) {
    if (tid < s) { s1[tid] += s1[tid + s]; s2[tid] += s2[tid + s]; }
    __syncthreads();
  }
  if (tid == 0) { float mean = s1[0] / M; stats[c] = mean; stats[C + c] = s2[0] / M - mean * mean; }
}

__global__ void bn_apply_kernel(float* __restrict__ x, const float* __restrict__ stats,
                                const float* __restrict__ g, const float* __restrict__ be,
                                long long total, int C) {
  long long t = (long long)blockIdx.x * 256 + threadIdx.x;
  if (t >= total) return;
  int c = (int)(t % C);
  float v = (x[t] - stats[c]) * rsqrtf(stats[C + c] + EPSV) * g[c] + be[c];
  x[t] = fmaxf(v, 0.f);
}

__global__ void maxk_kernel(const float* __restrict__ in, float* __restrict__ out,
                            int SB, int K, int C) {
  long long t = (long long)blockIdx.x * 256 + threadIdx.x;
  long long tot = (long long)SB * C;
  if (t >= tot) return;
  int c = (int)(t % C);
  long long bs = t / C;
  float m = -1e30f;
  for (int k = 0; k < K; k++) m = fmaxf(m, in[(bs * K + k) * (long long)C + c]);
  out[t] = m;
}

__global__ void copy_cols_kernel(const float* __restrict__ src, float* __restrict__ dst,
                                 int M, int Cs, int Cd, int off) {
  long long t = (long long)blockIdx.x * 256 + threadIdx.x;
  if (t >= (long long)M * Cs) return;
  int m = (int)(t / Cs), c = (int)(t % Cs);
  dst[(long long)m * Cd + off + c] = src[t];
}

__global__ void add_kernel(const float* __restrict__ a, const float* __restrict__ b,
                           float* __restrict__ y, long long n) {
  long long t = (long long)blockIdx.x * 256 + threadIdx.x;
  if (t < n) y[t] = a[t] + b[t];
}

__global__ void pos_encode_kernel(const float* __restrict__ xyz, float* __restrict__ pe, int rows) {
  int t = blockIdx.x * 256 + threadIdx.x;
  if (t >= rows) return;
  const float PI_ = 3.14159265358979f;
  float p[3] = { xyz[t * 3], xyz[t * 3 + 1], xyz[t * 3 + 2] };
  float* o = pe + (long long)t * 60;
  for (int d = 0; d < 3; d++)
    for (int f = 0; f < 10; f++) {
      float v = p[d] * PI_ * (float)(1 << f);
      o[d * 10 + f]      = sinf(v);
      o[30 + d * 10 + f] = cosf(v);
    }
}

__device__ __forceinline__ void cvt4(FragH& fr, int o, float4 f, float s) {
  fr.h[o + 0] = (_Float16)(f.x * s); fr.h[o + 1] = (_Float16)(f.y * s);
  fr.h[o + 2] = (_Float16)(f.z * s); fr.h[o + 3] = (_Float16)(f.w * s);
}

// ---------------------------------------------------------------------------
// Flash attention: grid (T/128, H, B), 8 waves/block, wave owns a 16-row
// q-tile. All waves share (b,h), so the 32x64 K and V f32 tiles are staged in
// LDS once per block per key-tile via async global->LDS.
// ---------------------------------------------------------------------------
__global__ __launch_bounds__(256) void attn_kernel(const float* __restrict__ Q,
                                                   const float* __restrict__ Km,
                                                   const float* __restrict__ V,
                                                   float* __restrict__ O, int T, int D) {
  __shared__ float    sKt[32][64];   // [key][hd]
  __shared__ float    sVt[32][64];   // [key][hd]
  __shared__ float    sS[8][16][33];
  __shared__ _Float16 sP[8][16][32];
  __shared__ float    sM[8][16], sL[8][16], sC[8][16];
  int tid = threadIdx.x, wid = tid >> 5, lane = tid & 31;
  int l15 = lane & 15, hi = (lane >> 4) & 1;
  int b = blockIdx.z, h = blockIdx.y;
  int row0 = (blockIdx.x * 8 + wid) * 16;
  const float scale = 0.125f;  // hd^-0.5, folded into Q fragments

  // per-thread staging slot: 32 keys x 8 segments of 8 floats
  int skey = tid >> 3, sseg = (tid & 7) * 8;

  FragH aq[2];
  {
    const float* qrow = Q + ((long long)(b * T + row0 + l15)) * D + h * HDIM;
#pragma unroll
    for (int c2 = 0; c2 < 2; c2++) {
      const float* qp = qrow + (hi ? 8 : 0) + c2 * 32;
      cvt4(aq[c2], 0,  *(const float4*)(qp),      scale);
      cvt4(aq[c2], 4,  *(const float4*)(qp + 4),  scale);
      cvt4(aq[c2], 8,  *(const float4*)(qp + 16), scale);
      cvt4(aq[c2], 12, *(const float4*)(qp + 20), scale);
    }
  }
  FragF o4[4];
#pragma unroll
  for (int i = 0; i < 4; i++)
#pragma unroll
    for (int r = 0; r < 8; r++) o4[i].f[r] = 0.f;
  if (!hi) { sM[wid][l15] = -1e30f; sL[wid][l15] = 0.f; }
  __syncthreads();

  for (int j0 = 0; j0 < T; j0 += 32) {
    {  // stage K/V tiles (shared by all 8 waves)
      const float* kg = Km + ((long long)(b * T + j0 + skey)) * D + h * HDIM + sseg;
      const float* vg = V  + ((long long)(b * T + j0 + skey)) * D + h * HDIM + sseg;
      async_cp16(kg,     &sKt[skey][sseg]);
      async_cp16(kg + 4, &sKt[skey][sseg + 4]);
      async_cp16(vg,     &sVt[skey][sseg]);
      async_cp16(vg + 4, &sVt[skey][sseg + 4]);
      async_fence_block();
    }
    FragF s[2];
#pragma unroll
    for (int i = 0; i < 2; i++)
#pragma unroll
      for (int r = 0; r < 8; r++) s[i].f[r] = 0.f;
#pragma unroll
    for (int ck = 0; ck < 2; ck++)
#pragma unroll
      for (int kc = 0; kc < 2; kc++) {
        FragH bk;
        const float* kp = &sKt[ck * 16 + l15][(hi ? 8 : 0) + kc * 32];
#pragma unroll
        for (int e = 0; e < 16; e++) bk.h[e] = (_Float16)kp[(e < 8) ? e : e + 8];
        s[ck].v = __builtin_amdgcn_wmma_f32_16x16x32_f16(
            false, aq[kc].v, false, bk.v, (short)0, s[ck].v, false, false);
      }
#pragma unroll
    for (int ck = 0; ck < 2; ck++)
#pragma unroll
      for (int r = 0; r < 8; r++)
        sS[wid][r + (hi ? 8 : 0)][ck * 16 + l15] = s[ck].f[r];
    __syncthreads();
    if (!hi) {  // lanes 0-15 each own one row
      int row = l15;
      float mold = sM[wid][row], lold = sL[wid][row];
      float mt = mold;
      for (int j = 0; j < 32; j++) mt = fmaxf(mt, sS[wid][row][j]);
      float corr = __expf(mold - mt);
      float ls = lold * corr;
      for (int j = 0; j < 32; j++) {
        float p = __expf(sS[wid][row][j] - mt);
        ls += p;
        sP[wid][row][j] = (_Float16)p;
      }
      sM[wid][row] = mt; sL[wid][row] = ls; sC[wid][row] = corr;
    }
    __syncthreads();
#pragma unroll
    for (int i = 0; i < 4; i++)
#pragma unroll
      for (int r = 0; r < 8; r++) o4[i].f[r] *= sC[wid][r + (hi ? 8 : 0)];
    FragH ap;
#pragma unroll
    for (int e = 0; e < 16; e++) {
      int key = ((e < 8) ? e : e + 8) + (hi ? 8 : 0);
      ap.h[e] = sP[wid][l15][key];
    }
#pragma unroll
    for (int ct = 0; ct < 4; ct++) {
      FragH bv;
#pragma unroll
      for (int e = 0; e < 16; e++)
        bv.h[e] = (_Float16)sVt[((e < 8) ? e : e + 8) + (hi ? 8 : 0)][ct * 16 + l15];
      o4[ct].v = __builtin_amdgcn_wmma_f32_16x16x32_f16(
          false, ap.v, false, bv.v, (short)0, o4[ct].v, false, false);
    }
    __syncthreads();
  }
#pragma unroll
  for (int ct = 0; ct < 4; ct++)
#pragma unroll
    for (int r = 0; r < 8; r++) {
      int row = r + (hi ? 8 : 0);
      float inv = 1.f / sL[wid][row];
      O[((long long)(b * T + row0 + row)) * D + h * HDIM + ct * 16 + l15] = o4[ct].f[r] * inv;
    }
}

// LayerNorm over last dim with optional residual (y = LN(x + res))
__global__ void ln_kernel(const float* __restrict__ x, const float* __restrict__ res,
                          const float* __restrict__ g, const float* __restrict__ be,
                          float* __restrict__ y, int C) {
  int row = blockIdx.x, tid = threadIdx.x;
  __shared__ float s1[256], s2[256];
  const float* xr = x + (long long)row * C;
  const float* rr = res ? res + (long long)row * C : nullptr;
  float a = 0.f, q = 0.f;
  for (int c = tid; c < C; c += 256) { float v = xr[c] + (rr ? rr[c] : 0.f); a += v; q += v * v; }
  s1[tid] = a; s2[tid] = q;
  __syncthreads();
  for (int s = 128; s > 0; s >>= 1) {
    if (tid < s) { s1[tid] += s1[tid + s]; s2[tid] += s2[tid + s]; }
    __syncthreads();
  }
  float mean = s1[0] / C;
  float inv = rsqrtf(s2[0] / C - mean * mean + EPSV);
  float* yr = y + (long long)row * C;
  for (int c = tid; c < C; c += 256) {
    float v = xr[c] + (rr ? rr[c] : 0.f);
    yr[c] = (v - mean) * inv * g[c] + be[c];
  }
}

__global__ void mean_tokens_kernel(const float* __restrict__ x, float* __restrict__ out,
                                   int T, int C) {
  int t = blockIdx.x * 256 + threadIdx.x;
  if (t >= BATCH * C) return;
  int b = t / C, c = t % C;
  float a = 0.f;
  for (int i = 0; i < T; i++) a += x[((long long)b * T + i) * C + c];
  out[t] = a / T;
}

__global__ void bn1d_relu_kernel(float* __restrict__ x, const float* __restrict__ g,
                                 const float* __restrict__ be, int M, int C) {
  int c = blockIdx.x * 256 + threadIdx.x;
  if (c >= C) return;
  float a = 0.f, q = 0.f;
  for (int m = 0; m < M; m++) { float v = x[m * C + c]; a += v; q += v * v; }
  float mean = a / M;
  float inv = rsqrtf(q / M - mean * mean + EPSV);
  for (int m = 0; m < M; m++) {
    float v = (x[m * C + c] - mean) * inv * g[c] + be[c];
    x[m * C + c] = fmaxf(v, 0.f);
  }
}

// ---------------------------------------------------------------------------
// Host-side orchestration
// ---------------------------------------------------------------------------
static inline dim3 g1(long long n) { return dim3((unsigned)((n + 255) / 256)); }

static void gemm(const float* A, const float* W, const float* bias, float* C,
                 long long M, int N, int K, int act, hipStream_t s) {
  dim3 grid((unsigned)((N + 127) / 128), (unsigned)((M + 63) / 64));
  bool aligned = (M % 64 == 0) && (N % 128 == 0) && (K % 32 == 0);
  if (aligned)
    gemm_kernel<false><<<grid, 256, 0, s>>>(A, W, bias, C, (int)M, N, K, act);
  else
    gemm_kernel<true><<<grid, 256, 0, s>>>(A, W, bias, C, (int)M, N, K, act);
}

static void bn_relu_ip(float* x, const float* g, const float* be, long long M, int C,
                       float* stats, hipStream_t s) {
  bn_stats_kernel<<<C, 256, 0, s>>>(x, stats, (int)M, C);
  long long tot = M * C;
  bn_apply_kernel<<<g1(tot), 256, 0, s>>>(x, stats, g, be, tot, C);
}

static void run_sa(const float* xyz, int N, const float* feats, int Cf, int S, float radius,
                   const float* const* P, const int* couts,
                   float* newxyz, float* newfeat, float* arenaA, float* arenaB,
                   int* fpsidx, int* ballidx, float* stats, hipStream_t s) {
  fps_kernel<<<BATCH, 256, (size_t)N * sizeof(float), s>>>(xyz, fpsidx, N, S);
  gather_kernel<<<g1((long long)BATCH * S * 3), 256, 0, s>>>(xyz, fpsidx, newxyz, N, S, 3);
  ball_query_kernel<<<g1((long long)BATCH * S), 256, 0, s>>>(xyz, newxyz, ballidx, N, S, 32,
                                                             radius * radius);
  group_kernel<<<g1((long long)BATCH * S * 32), 256, 0, s>>>(xyz, newxyz, feats, ballidx,
                                                             arenaA, N, S, 32, Cf);
  long long rows = (long long)BATCH * S * 32;
  int cin = 3 + Cf;
  float* x = arenaA; float* y = arenaB;
  for (int l = 0; l < 3; l++) {
    int co = couts[l];
    gemm(x, P[l * 4 + 0], P[l * 4 + 1], y, rows, co, cin, 0, s);
    bn_relu_ip(y, P[l * 4 + 2], P[l * 4 + 3], rows, co, stats, s);
    float* t = x; x = y; y = t; cin = co;
  }
  maxk_kernel<<<g1((long long)BATCH * S * couts[2]), 256, 0, s>>>(x, newfeat, BATCH * S, 32,
                                                                  couts[2]);
}

extern "C" void kernel_launch(void* const* d_in, const int* in_sizes, int n_in,
                              void* d_out, int out_size, void* d_ws, size_t ws_size,
                              hipStream_t stream) {
  (void)in_sizes; (void)n_in; (void)out_size; (void)ws_size;
  auto F = [&](int i) -> const float* { return (const float*)d_in[i]; };

  char* base = (char*)d_ws; size_t off = 0;
  auto alloc = [&](size_t bytes) -> void* {
    size_t o = (off + 255) & ~(size_t)255;
    void* p = base + o; off = o + bytes; return p;
  };
  auto fal = [&](long long n) -> float* { return (float*)alloc((size_t)n * 4); };
  auto ial = [&](long long n) -> int*   { return (int*)alloc((size_t)n * 4); };

  float* stats  = fal(2048);
  int*   fpsidx = ial((long long)BATCH * 2048);
  int*   ballidx= ial((long long)BATCH * 2048 * 32);
  float* l1x = fal((long long)BATCH * 2048 * 3);
  float* l1f = fal((long long)BATCH * 2048 * 128);
  float* l2x = fal((long long)BATCH * 512 * 3);
  float* l2f = fal((long long)BATCH * 512 * 256);
  float* l3x = fal((long long)BATCH * 128 * 3);
  float* l3f = fal((long long)BATCH * 128 * 512);
  float* arenaA = fal((long long)BATCH * 2048 * 32 * 128);
  float* arenaB = fal((long long)BATCH * 2048 * 32 * 128);
  float* pn = fal(BATCH * 1024);
  long long TD = (long long)BATCH * TOK * DMODEL;
  float* tA  = fal(TD); float* tB  = fal(TD); float* pos = fal(TD);
  float* qb  = fal(TD); float* Qb  = fal(TD); float* Kb  = fal(TD);
  float* Vb  = fal(TD); float* Ob  = fal(TD);
  float* ffn = fal((long long)BATCH * TOK * FFNDIM);
  float* pe  = fal((long long)BATCH * TOK * 60);
  float* tmean = fal(BATCH * DMODEL);
  float* tpool = fal(BATCH * DMODEL);
  float* tg = fal(BATCH * 1024);
  float* fv = fal(BATCH * 1024);
  float* fo = fal(BATCH * 1024);
  float* fused = fal(BATCH * 1024);
  float* cat2 = fal(BATCH * 2048);
  float* h1 = fal(BATCH * 1024);
  float* feat = fal(BATCH * 512);
  float* c1 = fal(BATCH * 256);
  float* c2 = fal(BATCH * 128);

  static const int saw[4][3] = {{1, 5, 9}, {13, 17, 21}, {25, 29, 33}, {37, 41, 45}};
  static const int sac[4][3] = {{64, 64, 128}, {128, 128, 256}, {256, 256, 512}, {512, 512, 1024}};
  const float* P[12];

  for (int l = 0; l < 3; l++) for (int k = 0; k < 4; k++) P[l * 4 + k] = F(saw[0][l] + k);
  run_sa(F(0), NPTS, nullptr, 0, 2048, 0.2f, P, sac[0], l1x, l1f, arenaA, arenaB,
         fpsidx, ballidx, stats, stream);
  for (int l = 0; l < 3; l++) for (int k = 0; k < 4; k++) P[l * 4 + k] = F(saw[1][l] + k);
  run_sa(l1x, 2048, l1f, 128, 512, 0.4f, P, sac[1], l2x, l2f, arenaA, arenaB,
         fpsidx, ballidx, stats, stream);
  for (int l = 0; l < 3; l++) for (int k = 0; k < 4; k++) P[l * 4 + k] = F(saw[2][l] + k);
  run_sa(l2x, 512, l2f, 256, 128, 0.8f, P, sac[2], l3x, l3f, arenaA, arenaB,
         fpsidx, ballidx, stats, stream);

  // SA4: global MLP over the 128 l3 points, then max-pool -> pn_global
  {
    copy_cols_kernel<<<g1((long long)BATCH * 128 * 3), 256, 0, stream>>>(l3x, arenaA,
                                                                         BATCH * 128, 3, 515, 0);
    copy_cols_kernel<<<g1((long long)BATCH * 128 * 512), 256, 0, stream>>>(l3f, arenaA,
                                                                           BATCH * 128, 512, 515, 3);
    long long rows = BATCH * 128;
    int cin = 515;
    float* x = arenaA; float* y = arenaB;
    for (int l = 0; l < 3; l++) {
      int w = saw[3][l], co = sac[3][l];
      gemm(x, F(w), F(w + 1), y, rows, co, cin, 0, stream);
      bn_relu_ip(y, F(w + 2), F(w + 3), rows, co, stats, stream);
      float* t = x; x = y; y = t; cin = co;
    }
    maxk_kernel<<<g1((long long)BATCH * 1024), 256, 0, stream>>>(x, pn, BATCH, 128, 1024);
  }

  // Transformer over l1 points (T=2048 tokens)
  long long RT = (long long)BATCH * TOK;
  gemm(l1x, F(51), F(52), tA, RT, DMODEL, 3, 0, stream);
  pos_encode_kernel<<<g1(RT), 256, 0, stream>>>(l1x, pe, (int)RT);
  gemm(pe, F(49), F(50), pos, RT, DMODEL, 60, 0, stream);
  float* cur = tA; float* nxt = tB;
  for (int l = 0; l < 4; l++) {
    int p0 = 53 + l * 12;
    add_kernel<<<g1(TD), 256, 0, stream>>>(cur, pos, qb, TD);
    gemm(qb,  F(p0),                                   F(p0 + 1),              Qb, RT, DMODEL, DMODEL, 0, stream);
    gemm(qb,  F(p0) + (long long)DMODEL * DMODEL,      F(p0 + 1) + DMODEL,     Kb, RT, DMODEL, DMODEL, 0, stream);
    gemm(cur, F(p0) + (long long)2 * DMODEL * DMODEL,  F(p0 + 1) + 2 * DMODEL, Vb, RT, DMODEL, DMODEL, 0, stream);
    attn_kernel<<<dim3(TOK / 128, NHEADS, BATCH), 256, 0, stream>>>(Qb, Kb, Vb, Ob, TOK, DMODEL);
    gemm(Ob, F(p0 + 2), F(p0 + 3), qb, RT, DMODEL, DMODEL, 0, stream);
    ln_kernel<<<(unsigned)RT, 256, 0, stream>>>(cur, qb, F(p0 + 8), F(p0 + 9), nxt, DMODEL);
    gemm(nxt, F(p0 + 4), F(p0 + 5), ffn, RT, FFNDIM, DMODEL, 1, stream);
    gemm(ffn, F(p0 + 6), F(p0 + 7), Qb, RT, DMODEL, FFNDIM, 0, stream);
    ln_kernel<<<(unsigned)RT, 256, 0, stream>>>(nxt, Qb, F(p0 + 10), F(p0 + 11), cur, DMODEL);
  }
  mean_tokens_kernel<<<g1(BATCH * DMODEL), 256, 0, stream>>>(cur, tmean, TOK, DMODEL);
  gemm(tmean, F(101), F(102), tpool, BATCH, DMODEL, DMODEL, 1, stream);
  gemm(tpool, F(103), F(104), tg, BATCH, 1024, DMODEL, 0, stream);

  // Fused single-key MHA: softmax over one key == 1, so out = (tg Wv^T+bv) Wo^T + bo
  gemm(tg, F(105) + (long long)2048 * 1024, F(106) + 2048, fv, BATCH, 1024, 1024, 0, stream);
  gemm(fv, F(107), F(108), fo, BATCH, 1024, 1024, 0, stream);
  ln_kernel<<<BATCH, 256, 0, stream>>>(fo, pn, F(109), F(110), fused, 1024);

  copy_cols_kernel<<<g1((long long)BATCH * 1024), 256, 0, stream>>>(fused, cat2, BATCH, 1024, 2048, 0);
  copy_cols_kernel<<<g1((long long)BATCH * 1024), 256, 0, stream>>>(tg,    cat2, BATCH, 1024, 2048, 1024);
  gemm(cat2, F(111), F(112), h1,   BATCH, 1024, 2048, 1, stream);
  gemm(h1,   F(113), F(114), feat, BATCH, 512,  1024, 0, stream);
  gemm(feat, F(115), F(116), c1,   BATCH, 256,  512,  0, stream);
  bn1d_relu_kernel<<<1, 256, 0, stream>>>(c1, F(117), F(118), BATCH, 256);
  gemm(c1, F(119), F(120), c2, BATCH, 128, 256, 0, stream);
  bn1d_relu_kernel<<<1, 256, 0, stream>>>(c2, F(121), F(122), BATCH, 128);
  gemm(c2, F(123), F(124), (float*)d_out, BATCH, 6, 128, 0, stream);
}
